// MyRNN_26130581029038
// MI455X (gfx1250) — compile-verified, matrix operand order checked by reading, not run on hardware
//
#include <hip/hip_runtime.h>

typedef __attribute__((ext_vector_type(16))) _Float16 v16h;
typedef __attribute__((ext_vector_type(8)))  _Float16 v8h;
typedef __attribute__((ext_vector_type(4)))  _Float16 v4h;
typedef __attribute__((ext_vector_type(8)))  float    v8f;

#define B_   512
#define T_   512
#define I_   128
#define HD_  256
#define C_   128

#define ROWS_PER_BLOCK 16
#define THREADS 512      // 16 wave32s
#define NWAVES  16

// numerically stable tanh via v_exp_f32 + rcp
__device__ __forceinline__ float fast_tanh(float x) {
    float ax = __builtin_fabsf(x);
    float e  = __expf(-2.0f * ax);
    float t  = (1.0f - e) / (1.0f + e);
    return __builtin_copysignf(t, x);
}

// Load a 16-bit B-operand fragment (32x16 K-slice starting at K0, column `col`)
// from a row-major f32 matrix W with leading dimension ldw, converting to f16.
// Layout mirrors the ISA 16-bit operand mapping:
//   lanes 0-15 : elems 0-7 -> K0+0..7,  elems 8-15 -> K0+16..23
//   lanes 16-31: elems 0-7 -> K0+8..15, elems 8-15 -> K0+24..31
__device__ __forceinline__ v16h load_bfrag_f32(const float* __restrict__ W, int ldw,
                                               int K0, int col, int lane) {
    const int hg = lane >> 4;
    v16h f;
#pragma unroll
    for (int j = 0; j < 8; ++j)
        f[j] = (_Float16)W[(size_t)(K0 + hg * 8 + j) * ldw + col];
#pragma unroll
    for (int j = 0; j < 8; ++j)
        f[8 + j] = (_Float16)W[(size_t)(K0 + 16 + hg * 8 + j) * ldw + col];
    return f;
}

// Load a 16-bit A-operand fragment (16 rows x 32 K, K base = K0) from an f16
// row-major LDS tile with leading dimension ld (in halves). Two 16B ds loads.
__device__ __forceinline__ v16h load_afrag_lds(const _Float16* S, int ld,
                                               int K0, int lane) {
    const int row = lane & 15;
    const int hg  = lane >> 4;
    const _Float16* p = S + row * ld + K0 + hg * 8;
    v8h lo = *(const v8h*)(p);        // K0+hg*8    .. +7
    v8h hi = *(const v8h*)(p + 16);   // K0+16+hg*8 .. +7
    return __builtin_shufflevector(lo, hi, 0,1,2,3,4,5,6,7,8,9,10,11,12,13,14,15);
}

__global__ __launch_bounds__(THREADS)
void rnn_fused_kernel(const float* __restrict__ X,      // [B,1,T,I]
                      const float* __restrict__ W_xh,   // [I,Hd]
                      const float* __restrict__ W_hh,   // [Hd,Hd]
                      const float* __restrict__ b_h,    // [Hd]
                      const float* __restrict__ W_hq,   // [Hd,C]
                      const float* __restrict__ b_q,    // [C]
                      float* __restrict__ out)          // [B,C]
{
    // Double-buffered H and X: write target is always disjoint from read
    // source, so ONE barrier per recurrence step is sufficient.
    __shared__ _Float16 Hlds[2][ROWS_PER_BLOCK * HD_];   // 2 x 16x256 f16 = 16 KB
    __shared__ _Float16 Xlds[2][ROWS_PER_BLOCK * I_];    // 2 x 16x128 f16 =  8 KB

    const int tid  = threadIdx.x;
    const int lane = tid & 31;            // wave32
    const int wave = tid >> 5;            // 0..15
    const int b0   = blockIdx.x * ROWS_PER_BLOCK;
    const int col  = wave * 16 + (lane & 15);   // this wave's Hd column
    const int hg   = lane >> 4;

    // ---- register-resident weight fragments (loaded once) ----
    v16h w_hh[8];                          // H @ W_hh, K = 256
#pragma unroll
    for (int kk = 0; kk < 8; ++kk)
        w_hh[kk] = load_bfrag_f32(W_hh, HD_, kk * 32, col, lane);
    v16h w_xh[4];                          // X @ W_xh, K = 128
#pragma unroll
    for (int kk = 0; kk < 4; ++kk)
        w_xh[kk] = load_bfrag_f32(W_xh, HD_, kk * 32, col, lane);

    const float bh = b_h[col];

    // ---- H_0 = 0 (only buffer 0 is read at t=0) ----
    for (int i = tid; i < ROWS_PER_BLOCK * HD_; i += THREADS)
        Hlds[0][i] = (_Float16)0.0f;

    // X staging plan: 16x128 f32 tile = 2048 floats; 512 threads x 4 floats.
    const int xIdx = tid * 4;
    const int xRow = xIdx >> 7;            // / 128
    const int xCol = xIdx & 127;
    const float* Xbase = X + ((size_t)(b0 + xRow) * T_) * I_ + xCol;
    _Float16* const xSlot0 = &Xlds[0][xRow * I_ + xCol];
    _Float16* const xSlot1 = &Xlds[1][xRow * I_ + xCol];

    // Prologue: stage X_0 into buffer 0, start the load of X_1.
    {
        const float4 xv = *(const float4*)(Xbase);
        v4h h4 = { (_Float16)xv.x, (_Float16)xv.y,
                   (_Float16)xv.z, (_Float16)xv.w };
        *(v4h*)(xSlot0) = h4;
    }
    float4 xreg = *(const float4*)(Xbase + (size_t)1 * I_);   // X_1 in flight

    __syncthreads();

    for (int t = 0; t < T_; ++t) {
        const _Float16* Hcur = Hlds[t & 1];
        const _Float16* Xcur = Xlds[t & 1];
        _Float16* Hnxt = Hlds[(t + 1) & 1];

        // ---- acc = b_h; acc += X_t @ W_xh + H_t @ W_hh ----
        v8f acc;
#pragma unroll
        for (int r = 0; r < 8; ++r) acc[r] = bh;

#pragma unroll
        for (int kk = 0; kk < 4; ++kk) {
            v16h a = load_afrag_lds(Xcur, I_, kk * 32, lane);
            acc = __builtin_amdgcn_wmma_f32_16x16x32_f16(
                      false, a, false, w_xh[kk], (short)0, acc, false, false);
        }
#pragma unroll
        for (int kk = 0; kk < 8; ++kk) {
            v16h a = load_afrag_lds(Hcur, HD_, kk * 32, lane);
            acc = __builtin_amdgcn_wmma_f32_16x16x32_f16(
                      false, a, false, w_hh[kk], (short)0, acc, false, false);
        }

        // ---- stage X_{t+1} (already in flight) into the other buffer,
        //      and launch the load for X_{t+2} (clamped at the end) ----
        if (t + 1 < T_) {
            v4h h4 = { (_Float16)xreg.x, (_Float16)xreg.y,
                       (_Float16)xreg.z, (_Float16)xreg.w };
            *(v4h*)(((t + 1) & 1) ? xSlot1 : xSlot0) = h4;
            const int tn = (t + 2 < T_) ? (t + 2) : (T_ - 1);
            xreg = *(const float4*)(Xbase + (size_t)tn * I_);
            __builtin_prefetch(Xbase + (size_t)((tn + 1 < T_) ? tn + 1 : tn) * I_, 0, 1);
        }

        // ---- H_{t+1} = tanh(acc), written to the other H buffer ----
#pragma unroll
        for (int r = 0; r < 8; ++r)
            Hnxt[(r + 8 * hg) * HD_ + col] = (_Float16)fast_tanh(acc[r]);

        __syncthreads();   // publish H_{t+1} and X_{t+1}; protects next overwrite
    }

    // ---- out = H_last @ W_hq + b_q  (C = 128 -> 8 column tiles) ----
    const _Float16* Hfin = Hlds[T_ & 1];
    if (wave < 8) {
        const int ccol = wave * 16 + (lane & 15);
        const float bq = b_q[ccol];
        v8f acc;
#pragma unroll
        for (int r = 0; r < 8; ++r) acc[r] = bq;
#pragma unroll
        for (int kk = 0; kk < 8; ++kk) {
            v16h a = load_afrag_lds(Hfin, HD_, kk * 32, lane);
            v16h b = load_bfrag_f32(W_hq, C_, kk * 32, ccol, lane);
            acc = __builtin_amdgcn_wmma_f32_16x16x32_f16(
                      false, a, false, b, (short)0, acc, false, false);
        }
#pragma unroll
        for (int r = 0; r < 8; ++r)
            out[(size_t)(b0 + r + 8 * hg) * C_ + ccol] = acc[r];
    }
}

extern "C" void kernel_launch(void* const* d_in, const int* in_sizes, int n_in,
                              void* d_out, int out_size, void* d_ws, size_t ws_size,
                              hipStream_t stream) {
    (void)in_sizes; (void)n_in; (void)out_size; (void)d_ws; (void)ws_size;
    const float* X    = (const float*)d_in[0];
    const float* W_xh = (const float*)d_in[1];
    const float* W_hh = (const float*)d_in[2];
    const float* b_h  = (const float*)d_in[3];
    const float* W_hq = (const float*)d_in[4];
    const float* b_q  = (const float*)d_in[5];
    float* out = (float*)d_out;

    dim3 grid(B_ / ROWS_PER_BLOCK);   // 32 blocks, each owns 16 batch rows
    dim3 block(THREADS);              // 16 wave32s
    rnn_fused_kernel<<<grid, block, 0, stream>>>(X, W_xh, W_hh, b_h, W_hq, b_q, out);
}
